// LSTM_6158983102526
// MI455X (gfx1250) — compile-verified, matrix operand order checked by reading, not run on hardware
//
#include <hip/hip_runtime.h>
#include <hip/hip_bf16.h>
#include <math.h>

typedef _Float16 v16h __attribute__((ext_vector_type(16)));
typedef _Float16 v8h  __attribute__((ext_vector_type(8)));
typedef float    v8f  __attribute__((ext_vector_type(8)));

#define B_   1024
#define T_   128
#define E_   300
#define EP_  320     // E padded to k-tile multiple (10 x 32)
#define H_   100
#define HP_  128     // H padded (4 x 32)
#define G4_  400     // 4*H, exactly 25 N-tiles of 16
#define NT_  25
#define KTX_ 10
#define KTH_ 4
#define V_   50000
#define NEG_INF_ (-1e30f)

// K index inside a 16x32 f16 A/B fragment for element e (0..15) in lane-half `half`
// (ISA 7.12.2: VGPR0..3 hold K runs of 8, VGPR4..7 the next runs)
__device__ __forceinline__ int kmap(int e, int half) {
  return (e & 7) + 8 * (((e >> 3) << 1) | half);
}

// ---- one-time prep kernels -------------------------------------------------

__global__ void pack_emb(const float* __restrict__ emb, _Float16* __restrict__ dst, long n) {
  long i = (long)blockIdx.x * blockDim.x + threadIdx.x;
  if (i >= n) return;
  int  k = (int)(i % EP_);
  long v = i / EP_;
  dst[i] = (k < E_) ? (_Float16)emb[v * (long)E_ + k] : (_Float16)0.f;
}

// Pack row-major W[n, k] (gates = x @ W^T) into per-lane WMMA B fragments:
// dst[((nt*nkt + kt)*32 + lane)*16 + e] = W[nt*16 + (lane&15), kt*32 + kmap(e, lane>>4)]
__global__ void pack_w(const float* __restrict__ W, _Float16* __restrict__ dst,
                       int nkt, int kdim) {
  int i = blockIdx.x * blockDim.x + threadIdx.x;
  int total = NT_ * nkt * 32 * 16;
  if (i >= total) return;
  int e    = i & 15;
  int lane = (i >> 4) & 31;
  int q    = i >> 9;           // nt*nkt + kt
  int kt   = q % nkt;
  int nt   = q / nkt;
  int half = lane >> 4;
  int n = nt * 16 + (lane & 15);
  int k = kt * 32 + kmap(e, half);
  dst[i] = (k < kdim) ? (_Float16)W[n * kdim + k] : (_Float16)0.f;
}

__global__ void bias_kernel(const float* __restrict__ bi, const float* __restrict__ bh,
                            float* __restrict__ dst) {
  int i = blockIdx.x * blockDim.x + threadIdx.x;
  if (i < G4_) dst[i] = bi[i] + bh[i];
}

// ---- fused LSTM: 16 batch rows per block, private loop over all T steps ----

__global__ void __launch_bounds__(128)
lstm_kernel(const int* __restrict__ word_id,
            const _Float16* __restrict__ emb16,
            const _Float16* __restrict__ wih,   // packed B fragments
            const _Float16* __restrict__ whh,   // packed B fragments
            const float* __restrict__ bias,
            float* __restrict__ hout)           // (B,T,H) f32
{
  __shared__ _Float16 sX[16][EP_];   // x_t tile, f16, zero-padded cols
  __shared__ _Float16 sH[16][HP_];   // h_{t-1} tile, f16, zero-padded cols
  __shared__ float    sG[16][G4_];   // gate accumulators (f32 from WMMA)
  __shared__ float    sC[16][H_];    // cell state
  __shared__ float    sB[G4_];       // combined bias
  __shared__ int      sWid[16][T_];  // this block's word ids

  const int tid  = threadIdx.x;
  const int lane = tid & 31;
  const int wave = tid >> 5;
  const int b0   = blockIdx.x * 16;

  for (int i = tid; i < 16 * T_; i += 128)
    sWid[i / T_][i % T_] = word_id[(b0 + i / T_) * T_ + (i % T_)];
  for (int i = tid; i < G4_; i += 128) sB[i] = bias[i];
  for (int i = tid; i < 16 * HP_; i += 128) sH[i / HP_][i % HP_] = (_Float16)0.f;
  for (int i = tid; i < 16 * H_; i += 128) sC[i / H_][i % H_] = 0.f;
  __syncthreads();

  const int half = lane >> 4;
  const int col  = lane & 15;
  const int mrow = lane & 15;

  for (int t = 0; t < T_; ++t) {
    // stage x_t: gather 16 embedding rows (f16, padded) into LDS, 16B vectors
    for (int c = tid; c < 16 * (EP_ / 8); c += 128) {
      int m  = c / (EP_ / 8);
      int kb = (c % (EP_ / 8)) * 8;
      *(v8h*)&sX[m][kb] = *(const v8h*)(emb16 + (long)sWid[m][t] * EP_ + kb);
    }
    if (t + 1 < T_) {
      int m = tid & 15;
      __builtin_prefetch(emb16 + (long)sWid[m][t + 1] * EP_, 0, 1);  // global_prefetch
    }
    __syncthreads();

    // gates(16 x 400) = x_t(16x320) @ Wih + h(16x128) @ Whh   via WMMA f16->f32
    for (int nt = wave; nt < NT_; nt += 4) {
      v8f acc = {};
      for (int kt = 0; kt < KTX_; ++kt) {
        const _Float16* p = &sX[mrow][kt * 32 + 8 * half];
        v8h lo = *(const v8h*)p;          // ds_load_b128
        v8h hi = *(const v8h*)(p + 16);   // ds_load_b128
        v16h a;
        #pragma unroll
        for (int e = 0; e < 8; ++e) { a[e] = lo[e]; a[e + 8] = hi[e]; }
        v16h b = *(const v16h*)(wih + ((size_t)(nt * KTX_ + kt) * 32 + lane) * 16);
        acc = __builtin_amdgcn_wmma_f32_16x16x32_f16(false, a, false, b,
                                                     (short)0, acc, false, false);
      }
      for (int kt = 0; kt < KTH_; ++kt) {
        const _Float16* p = &sH[mrow][kt * 32 + 8 * half];
        v8h lo = *(const v8h*)p;
        v8h hi = *(const v8h*)(p + 16);
        v16h a;
        #pragma unroll
        for (int e = 0; e < 8; ++e) { a[e] = lo[e]; a[e + 8] = hi[e]; }
        v16h b = *(const v16h*)(whh + ((size_t)(nt * KTH_ + kt) * 32 + lane) * 16);
        acc = __builtin_amdgcn_wmma_f32_16x16x32_f16(false, a, false, b,
                                                     (short)0, acc, false, false);
      }
      #pragma unroll
      for (int r = 0; r < 8; ++r)          // C/D layout: M = r + 8*half, N = col
        sG[r + 8 * half][nt * 16 + col] = acc[r];
    }
    __syncthreads();

    // elementwise gate math: i,f,g,o -> c,h
    for (int i = tid; i < 16 * H_; i += 128) {
      int m = i & 15, j = i >> 4;
      float gi = sG[m][j]       + sB[j];
      float gf = sG[m][j + 100] + sB[j + 100];
      float gg = sG[m][j + 200] + sB[j + 200];
      float go = sG[m][j + 300] + sB[j + 300];
      float si = 1.f / (1.f + __expf(-gi));
      float sf = 1.f / (1.f + __expf(-gf));
      float so = 1.f / (1.f + __expf(-go));
      float c  = sf * sC[m][j] + si * tanhf(gg);
      float h  = so * tanhf(c);
      sC[m][j] = c;
      sH[m][j] = (_Float16)h;
      hout[((size_t)(b0 + m) * T_ + t) * H_ + j] = h;
    }
    __syncthreads();
  }
}

// ---- heads: out_f, scores, top-k, pos/l_rep, per-row neg sums --------------

__global__ void __launch_bounds__(128)
finish1(const int* __restrict__ sen_len,
        const int* __restrict__ label_word_id,
        const float* __restrict__ emb,
        const float* __restrict__ lin_W, const float* __restrict__ lin_b,
        const float* __restrict__ ll_W,  const float* __restrict__ ll_b,
        const float* __restrict__ hout,
        float* __restrict__ per_neg,
        float* __restrict__ out)
{
  __shared__ float sLab[H_];
  __shared__ float sScore[T_];
  __shared__ float sPos[H_];
  __shared__ int   sIdx[4];
  __shared__ float sVal[4];

  const int b = blockIdx.x, tid = threadIdx.x;
  const int L = sen_len[b];
  const float* hb = hout + (size_t)b * T_ * H_;

  if (tid < H_) {               // label_vec = emb[label] @ ll_W^T + ll_b
    int wid = label_word_id[b];
    float acc = ll_b[tid];
    const float* er = emb + (size_t)wid * E_;
    for (int e = 0; e < E_; ++e) acc += er[e] * ll_W[tid * E_ + e];
    sLab[tid] = acc;
  }
  __syncthreads();

  if (tid < 6) {                // out_f = last_hidden @ lin_W^T + lin_b
    const float* last = hb + (size_t)(L - 1) * H_;
    float acc = lin_b[tid];
    for (int j = 0; j < H_; ++j) acc += last[j] * lin_W[tid * H_ + j];
    out[b * 6 + tid] = acc;
  }

  {                             // scores[t] = <h_t, label_vec> (masked)
    float s = NEG_INF_;
    if (tid < L) {
      s = 0.f;
      for (int j = 0; j < H_; ++j) s += hb[(size_t)tid * H_ + j] * sLab[j];
    }
    sScore[tid] = s;
  }
  __syncthreads();

  if (tid == 0) {               // top-4 (serial: T=128 only)
    for (int k = 0; k < 4; ++k) {
      int am = 0; float v = sScore[0];
      for (int t2 = 1; t2 < T_; ++t2) if (sScore[t2] > v) { v = sScore[t2]; am = t2; }
      sIdx[k] = am; sVal[k] = v; sScore[am] = -3.0e38f;
    }
  }
  __syncthreads();

  if (tid < H_) {
    float pos = 0.f, neg = 0.f;
    for (int k = 0; k < 4; ++k) pos += sVal[k] * hb[(size_t)sIdx[k] * H_ + tid];
    for (int t2 = 0; t2 < L; ++t2) {
      bool skip = (t2 == sIdx[0]) | (t2 == sIdx[1]) | (t2 == sIdx[2]) | (t2 == sIdx[3]);
      if (!skip) neg += hb[(size_t)t2 * H_ + tid];
    }
    sPos[tid] = pos;
    per_neg[(size_t)b * H_ + tid] = neg;
  }
  __syncthreads();

  if (tid < 6) {                // l_rep
    float acc = lin_b[tid];
    for (int j = 0; j < H_; ++j) acc += sPos[j] * lin_W[tid * H_ + j];
    out[B_ * 6 + b * 6 + tid] = acc;
  }
}

// cumsum over batch of per_neg, then r_rep (sequential dependency -> 1 block)
__global__ void __launch_bounds__(128)
finish2(const float* __restrict__ per_neg,
        const float* __restrict__ lin_W, const float* __restrict__ lin_b,
        float* __restrict__ out)
{
  __shared__ float sAcc[H_];
  const int tid = threadIdx.x;
  float acc = 0.f;
  for (int b = 0; b < B_; ++b) {
    if (tid < H_) { acc += per_neg[(size_t)b * H_ + tid]; sAcc[tid] = acc; }
    __syncthreads();
    if (tid < 6) {
      float r = lin_b[tid];
      for (int j = 0; j < H_; ++j) r += sAcc[j] * lin_W[tid * H_ + j];
      out[2 * B_ * 6 + b * 6 + tid] = r;
    }
    __syncthreads();
  }
}

// ---- launch ----------------------------------------------------------------

extern "C" void kernel_launch(void* const* d_in, const int* in_sizes, int n_in,
                              void* d_out, int out_size, void* d_ws, size_t ws_size,
                              hipStream_t stream) {
  const int*   word_id = (const int*)d_in[0];
  const int*   sen_len = (const int*)d_in[1];
  const int*   label   = (const int*)d_in[2];
  const float* emb     = (const float*)d_in[3];
  const float* W_ih    = (const float*)d_in[4];
  const float* W_hh    = (const float*)d_in[5];
  const float* b_ih    = (const float*)d_in[6];
  const float* b_hh    = (const float*)d_in[7];
  const float* lin_W   = (const float*)d_in[8];
  const float* lin_b   = (const float*)d_in[9];
  const float* ll_W    = (const float*)d_in[10];
  const float* ll_b    = (const float*)d_in[11];
  float* out = (float*)d_out;

  char* ws = (char*)d_ws;
  size_t off = 0;
  auto take = [&](size_t bytes) { size_t o = off; off += (bytes + 255) & ~(size_t)255; return o; };
  _Float16* emb16 = (_Float16*)(ws + take((size_t)V_ * EP_ * 2));          // 32 MB (L2-resident)
  _Float16* wihP  = (_Float16*)(ws + take((size_t)NT_ * KTX_ * 512 * 2));  // 256 KB
  _Float16* whhP  = (_Float16*)(ws + take((size_t)NT_ * KTH_ * 512 * 2));  // 100 KB
  float*    bias  = (float*)(ws + take((size_t)G4_ * 4));
  float*    hout  = (float*)(ws + take((size_t)B_ * T_ * H_ * 4));         // 52 MB
  float*    pneg  = (float*)(ws + take((size_t)B_ * H_ * 4));

  {
    long n = (long)V_ * EP_;
    pack_emb<<<(int)((n + 255) / 256), 256, 0, stream>>>(emb, emb16, n);
  }
  pack_w<<<(NT_ * KTX_ * 512 + 255) / 256, 256, 0, stream>>>(W_ih, wihP, KTX_, E_);
  pack_w<<<(NT_ * KTH_ * 512 + 255) / 256, 256, 0, stream>>>(W_hh, whhP, KTH_, H_);
  bias_kernel<<<2, 256, 0, stream>>>(b_ih, b_hh, bias);

  lstm_kernel<<<B_ / 16, 128, 0, stream>>>(word_id, emb16, wihP, whhP, bias, hout);

  finish1<<<B_, 128, 0, stream>>>(sen_len, label, emb, lin_W, lin_b, ll_W, ll_b,
                                  hout, pneg, out);
  finish2<<<1, 128, 0, stream>>>(pneg, lin_W, lin_b, out);
}
